// PortfolioModel_58471684767980
// MI455X (gfx1250) — compile-verified, matrix operand order checked by reading, not run on hardware
//
#include <hip/hip_runtime.h>
#include <hip/hip_bf16.h>

typedef __attribute__((ext_vector_type(16))) _Float16 v16h;
typedef __attribute__((ext_vector_type(8)))  float    v8f;

#define LBv 0.0f
#define UBv 0.5f
#define RFv 1.03f
#define ALPHAv 0.1f
#define OMEGAv 0.05f
#define BETAv 0.05f
#define MUv 0.01f
#define NWAVES 8   // 256 threads / wave32

// gfx1250 native V_TANH_F32 (confirmed lowering in round 2).
#if __has_builtin(__builtin_amdgcn_tanhf)
__device__ __forceinline__ float fast_tanh(float x) { return __builtin_amdgcn_tanhf(x); }
#elif __has_builtin(__builtin_amdgcn_tanh_f32)
__device__ __forceinline__ float fast_tanh(float x) { return __builtin_amdgcn_tanh_f32(x); }
#else
__device__ __forceinline__ float fast_tanh(float x) { return tanhf(x); }
#endif

// Raw v_exp_f32 (2^x). Softmax args are in [-2,0] -> no range reduction needed.
#if __has_builtin(__builtin_amdgcn_exp2f)
__device__ __forceinline__ float fast_exp(float x) {
    return __builtin_amdgcn_exp2f(x * 1.44269504088896340736f);
}
#else
__device__ __forceinline__ float fast_exp(float x) { return __expf(x); }
#endif

// Single v_rcp_f32 instead of a ~10-op precise divide sequence.
#if __has_builtin(__builtin_amdgcn_rcpf)
__device__ __forceinline__ float fast_rcp(float x) { return __builtin_amdgcn_rcpf(x); }
#else
__device__ __forceinline__ float fast_rcp(float x) { return 1.0f / x; }
#endif

// Async HBM<->LDS path (GLOBAL_LOAD_ASYNC_TO_LDS_B32 / GLOBAL_STORE_ASYNC_FROM_LDS_B32,
// ASYNCcnt-tracked). Per the ISA, INST_OFFSET is added to BOTH the global and LDS
// addresses, so a whole strided batch shares one global-base + one LDS-base VGPR
// with the stride folded into the immediate offset (literal ICE via repeat macros).
#if defined(__gfx1250__) &&                                              \
    __has_builtin(__builtin_amdgcn_global_load_async_to_lds_b32) &&      \
    __has_builtin(__builtin_amdgcn_global_store_async_from_lds_b32) &&   \
    __has_builtin(__builtin_amdgcn_s_wait_asynccnt)
#define ASYNC_LDS 1
#define GASYNC_LOAD_OFF(gbase, lbase, OFF)                               \
    __builtin_amdgcn_global_load_async_to_lds_b32(                       \
        (__attribute__((address_space(1))) int*)(gbase),                 \
        (__attribute__((address_space(3))) int*)(lbase), (OFF), 0)
#define GASYNC_STORE_OFF(gbase, lbase, OFF)                              \
    __builtin_amdgcn_global_store_async_from_lds_b32(                    \
        (__attribute__((address_space(1))) int*)(gbase),                 \
        (__attribute__((address_space(3))) int*)(lbase), (OFF), 0)
#define REP5(M)  M(0) M(1) M(2) M(3) M(4)
#define REP25(M) M(0)  M(1)  M(2)  M(3)  M(4)  M(5)  M(6)  M(7)  M(8)  M(9)  \
                 M(10) M(11) M(12) M(13) M(14) M(15) M(16) M(17) M(18) M(19) \
                 M(20) M(21) M(22) M(23) M(24)
#else
#define ASYNC_LDS 0
#endif

__global__ __launch_bounds__(256) void portfolio_k3_wmma_kernel(
    const float* __restrict__ x1,     // (B,)
    const float* __restrict__ x2,     // (B,5,5) row-major
    const float* __restrict__ eps3,   // (B,5)  == eps[3]
    const float* __restrict__ W3,     // (3,5,5) == W[3]
    float* __restrict__ outWealth,    // (B,)
    float* __restrict__ outCov)       // (B,5,5)
{
    __shared__ float stage[NWAVES][800];        // x2 stage in, cov stage out (reused)
    __shared__ float epss [NWAVES][160];        // eps stage
    __shared__ float hbuf [NWAVES][32][8];      // per-sample activations (padded rows)
    __shared__ float cbuf [NWAVES][2][16][16];  // WMMA D redistribution

    const int lane  = threadIdx.x & 31;
    const int wave  = threadIdx.x >> 5;
    const long sbase = (long)blockIdx.x * 256 + wave * 32;   // first sample of this wave
    const long s     = sbase + lane;                          // this lane's sample

    // ---- coalesced staging of x2/eps rows (HBM -> LDS, async; 128B/wave/op) ----
#if ASYNC_LDS
    {
        const float* gx2 = x2 + sbase * 25 + lane;
        const float* gep = eps3 + sbase * 5 + lane;
        float* lst = &stage[wave][lane];
        float* lep = &epss[wave][lane];
#define LD_X2(i) GASYNC_LOAD_OFF(gx2, lst, (i) * 128);
#define LD_EP(i) GASYNC_LOAD_OFF(gep, lep, (i) * 128);
        REP25(LD_X2)
        REP5(LD_EP)
#undef LD_X2
#undef LD_EP
    }
    __builtin_amdgcn_s_wait_asynccnt(0);
#else
    #pragma unroll
    for (int i = 0; i < 25; ++i)
        stage[wave][i * 32 + lane] = x2[sbase * 25 + i * 32 + lane];
    #pragma unroll
    for (int i = 0; i < 5; ++i)
        epss[wave][i * 32 + lane] = eps3[sbase * 5 + i * 32 + lane];
#endif
    __syncthreads();

    float xr[25], ev[5];
    #pragma unroll
    for (int i = 0; i < 25; ++i) xr[i] = stage[wave][lane * 25 + i];
    #pragma unroll
    for (int j = 0; j < 5; ++j)  ev[j] = epss[wave][lane * 5 + j];

    // ---- Ce = x2 @ eps (per-sample 5x5 matvec, VALU) ----
    float Ce[5];
    #pragma unroll
    for (int i = 0; i < 5; ++i) {
        float acc = 0.f;
        #pragma unroll
        for (int j = 0; j < 5; ++j) acc = fmaf(xr[i * 5 + j], ev[j], acc);
        Ce[i] = acc;
    }
    __syncthreads();   // everyone has read 'stage'; safe to overwrite with cov

    // ---- curr_cov[i][j] = OMEGA + ALPHA*x2[i][j] + BETA*Ce[j]^2 ----
    #pragma unroll
    for (int i = 0; i < 5; ++i)
        #pragma unroll
        for (int j = 0; j < 5; ++j)
            stage[wave][lane * 25 + i * 5 + j] =
                OMEGAv + ALPHAv * xr[i * 5 + j] + BETAv * Ce[j] * Ce[j];
    __syncthreads();
    // ---- coalesced cov store (LDS -> HBM, async; S_ENDPGM drains ASYNCcnt) ----
#if ASYNC_LDS
    {
        float* gcv = outCov + sbase * 25 + lane;
        float* lst = &stage[wave][lane];
#define ST_CV(i) GASYNC_STORE_OFF(gcv, lst, (i) * 128);
        REP25(ST_CV)
#undef ST_CV
    }
#else
    #pragma unroll
    for (int i = 0; i < 25; ++i)
        outCov[sbase * 25 + i * 32 + lane] = stage[wave][i * 32 + lane];
#endif

    // ---- R = MU + Ce ; h = R ----
    float R[5], h[5];
    #pragma unroll
    for (int i = 0; i < 5; ++i) { R[i] = MUv + Ce[i]; h[i] = R[i]; }

    // ---- 3 bias-free Linear(5,5)+tanh layers on the matrix pipe ----
    // Wave owns 32 samples = two 16-row WMMA tiles; N,K padded 5 -> 16/32.
    #pragma unroll
    for (int l = 0; l < 3; ++l) {
        #pragma unroll
        for (int e = 0; e < 5; ++e) hbuf[wave][lane][e] = h[e];
        __syncthreads();

        // A fragments (f16): lanes 0-15 carry K=0..7 (we use 0..4); lanes 16-31 = K pad (zero)
        v16h a0 = {}; v16h a1 = {};
        if (lane < 16) {
            #pragma unroll
            for (int e = 0; e < 5; ++e) {
                a0[e] = (_Float16)hbuf[wave][lane][e];        // tile0: samples 0..15
                a1[e] = (_Float16)hbuf[wave][lane + 16][e];   // tile1: samples 16..31
            }
        }
        // B fragment: lane n<5 holds W[n][0..4] as K=0..4 of column n, rest zero; out[n] = sum_k h[k]*W[n][k]
        v16h bf = {};
        if (lane < 5) {
            #pragma unroll
            for (int e = 0; e < 5; ++e)
                bf[e] = (_Float16)W3[l * 25 + lane * 5 + e];
        }

        v8f cz = {};
        v8f d0 = __builtin_amdgcn_wmma_f32_16x16x32_f16(false, a0, false, bf,
                                                        (short)0, cz, false, false);
        v8f d1 = __builtin_amdgcn_wmma_f32_16x16x32_f16(false, a1, false, bf,
                                                        (short)0, cz, false, false);

        // D layout: VGPR r, lanes 0-15 -> M=r, lanes 16-31 -> M=8+r; N = lane&15.
        // Only columns 0..4 are meaningful -> store just those.
        const int half = lane >> 4;
        const int col  = lane & 15;
        if (col < 5) {
            #pragma unroll
            for (int r = 0; r < 8; ++r) {
                cbuf[wave][0][half * 8 + r][col] = d0[r];
                cbuf[wave][1][half * 8 + r][col] = d1[r];
            }
        }
        __syncthreads();

        const int tile = lane >> 4;
        const int row  = lane & 15;
        #pragma unroll
        for (int n = 0; n < 5; ++n)
            h[n] = fast_tanh(cbuf[wave][tile][row][n]);
        __syncthreads();
    }

    // ---- softmax over 5 (args bounded: h in [-1,1]) ----
    float mx = h[0];
    #pragma unroll
    for (int i = 1; i < 5; ++i) mx = fmaxf(mx, h[i]);
    float ex[5], ssum = 0.f;
    #pragma unroll
    for (int i = 0; i < 5; ++i) { ex[i] = fast_exp(h[i] - mx); ssum += ex[i]; }
    const float rs = fast_rcp(ssum);
    float w[5];
    #pragma unroll
    for (int i = 0; i < 5; ++i) w[i] = ex[i] * rs;

    // ---- rebalance: exact replica of the 8-step scan semantics ----
    float oldv[5], newv[5];
    bool done = false;
    #pragma unroll
    for (int i = 0; i < 5; ++i) {
        oldv[i] = w[i];
        newv[i] = fminf(fmaxf(w[i], LBv), UBv);
    }
    #pragma unroll
    for (int it = 0; it < 8; ++it) {
        float leftover = 0.f, msum = 0.f, masked[5];
        #pragma unroll
        for (int i = 0; i < 5; ++i) leftover += oldv[i] - newv[i];
        #pragma unroll
        for (int i = 0; i < 5; ++i) {
            masked[i] = (newv[i] != UBv) ? newv[i] : 0.f;
            msum += masked[i];
        }
        const float rmsum = fast_rcp(msum);
        const float scale = leftover * rmsum;
        float new2[5];
        bool cont = false;
        #pragma unroll
        for (int i = 0; i < 5; ++i) {
            new2[i] = fmaf(scale, masked[i], newv[i]);
            cont = cont || (new2[i] > UBv);
        }
        #pragma unroll
        for (int i = 0; i < 5; ++i) {
            float new3 = cont ? fminf(fmaxf(new2[i], LBv), UBv) : new2[i];
            oldv[i] = done ? oldv[i] : new2[i];
            newv[i] = done ? newv[i] : new3;
        }
        done = done || !cont;
    }

    // ---- wealth = sum(x1 * w * (1+R) * RF) ----
    const float x1v = x1[s];
    float wealth = 0.f;
    #pragma unroll
    for (int i = 0; i < 5; ++i)
        wealth = fmaf(x1v * newv[i], (1.f + R[i]) * RFv, wealth);
    outWealth[s] = wealth;
}

extern "C" void kernel_launch(void* const* d_in, const int* in_sizes, int n_in,
                              void* d_out, int out_size, void* d_ws, size_t ws_size,
                              hipStream_t stream) {
    const float* x1  = (const float*)d_in[0];          // (B,)
    const float* x2  = (const float*)d_in[1];          // (B,5,5)
    const float* eps = (const float*)d_in[2];          // (4,B,5)
    const float* W   = (const float*)d_in[3];          // (4,3,5,5)
    const int B = in_sizes[0];

    // Only layer k = 3 reaches the outputs (reference overwrites each iter).
    const float* eps3 = eps + (size_t)3 * B * 5;
    const float* W3   = W + 9 * 25;                    // W[3,0,:,:]

    float* outW = (float*)d_out;
    float* outC = (float*)d_out + B;

    const int blocks = B / 256;                        // B = 524288 -> 2048
    portfolio_k3_wmma_kernel<<<blocks, 256, 0, stream>>>(x1, x2, eps3, W3, outW, outC);
}
